// PointNetPPLiteSeg_27547920236661
// MI455X (gfx1250) — compile-verified, matrix operand order checked by reading, not run on hardware
//
#include <hip/hip_runtime.h>

typedef __attribute__((ext_vector_type(16))) _Float16 v16h;
typedef __attribute__((ext_vector_type(8)))  float    v8f;
typedef _Float16 h16;

#define B_    8
#define N_    16384
#define S1_   1024
#define S2_   256
#define KNN_  32
#define NCLS_ 13

// ---------------------------------------------------------------------------
// Fragment-major ("swizzled") layouts so every WMMA operand is a contiguous
// 32-byte per-lane load (2 x b128) instead of 16 scattered u16 loads.
//
// A-fragment element (m, k) for k-step kt (koff = k - 32*kt):
//   koff in [0,16):  hi = koff>>3,        e = koff&7
//   koff in [16,32): hi = (koff-16)>>3,   e = 8 + ((koff-16)&7)
//   lane = m + 16*hi,  slot = (kt*32 + lane)*16 + e
// B tile (kt, nt): lane = (n&15) + 16*hi, same e mapping, stored at
//   ((kt*NT + nt)*32 + lane)*16 + e
// ---------------------------------------------------------------------------
__device__ __forceinline__ void stage_swz(h16* wbase, int m, int c, h16 v) {
  const int kt = c >> 5, koff = c & 31;
  const int hi = (koff < 16) ? (koff >> 3) : ((koff - 16) >> 3);
  const int e  = (koff < 16) ? (koff & 7) : (8 + ((koff - 16) & 7));
  wbase[((size_t)kt * 32 + (m + 16 * hi)) * 16 + e] = v;
}

// One wave: D = relu(A(16xKP) @ W(KPxNOUT) + bias).
// Aswz: wave-local swizzled A (KP*16 halves). Wpk: packed weights.
// SWZ=true -> write output into swizzled layout (feeds next layer's A);
// SWZ=false -> row-major 16 x NOUT.
template <int KP, int NOUT, bool SWZ>
__device__ __forceinline__ void wave_gemm_relu(const h16* __restrict__ Aswz,
                                               const h16* __restrict__ Wpk,
                                               const float* __restrict__ bias,
                                               h16* __restrict__ out) {
  constexpr int NT = NOUT / 16;
  constexpr int KT = KP / 32;
  const int lane = threadIdx.x & 31;
  v8f acc[NT];
#pragma unroll
  for (int t = 0; t < NT; ++t) acc[t] = (v8f){0.f,0.f,0.f,0.f,0.f,0.f,0.f,0.f};
#pragma unroll
  for (int kt = 0; kt < KT; ++kt) {
    const v16h a = *(const v16h*)(Aswz + ((size_t)kt * 32 + lane) * 16);
#pragma unroll
    for (int t = 0; t < NT; ++t) {
      const v16h b = *(const v16h*)(Wpk + (((size_t)kt * NT + t) * 32 + lane) * 16);
      acc[t] = __builtin_amdgcn_wmma_f32_16x16x32_f16(false, a, false, b,
                                                      (short)0, acc[t], false, false);
    }
  }
  const int hi = lane >> 4, nl = lane & 15;
#pragma unroll
  for (int t = 0; t < NT; ++t) {
#pragma unroll
    for (int r = 0; r < 8; ++r) {
      float v = acc[t][r] + bias[t * 16 + nl];
      v = v > 0.f ? v : 0.f;
      const int m = r + 8 * hi;
      if (SWZ) stage_swz(out, m, t * 16 + nl, (h16)v);
      else     out[m * NOUT + t * 16 + nl] = (h16)v;
    }
  }
}

// ---------------------------------------------------------------------------
// Weight convert: f32 [K][N] -> f16 fragment-major packed, K zero-padded.
// ---------------------------------------------------------------------------
__global__ void k_cvt_pack(const float* __restrict__ w, h16* __restrict__ o,
                           int K, int N, int Kpad) {
  const int i = blockIdx.x * 256 + threadIdx.x;
  if (i >= Kpad * N) return;
  const int e    = i & 15;
  const int lane = (i >> 4) & 31;
  const int tile = i >> 9;          // kt*NT + nt
  const int NT   = N >> 4;
  const int nt = tile % NT, kt = tile / NT;
  const int n  = nt * 16 + (lane & 15);
  const int hi = lane >> 4;
  const int koff = (e < 8) ? (hi * 8 + e) : (8 + hi * 8 + e); // 16 + hi*8 + (e-8)
  const int k = kt * 32 + koff;
  o[i] = (k < K) ? (h16)w[(size_t)k * N + n] : (h16)0.f;
}

// ---------------------------------------------------------------------------
// Embed: xyz = x[:, :3]; feat0 = relu(x @ ew + eb)  (6 -> 64, VALU)
// ---------------------------------------------------------------------------
__global__ void k_embed(const float* __restrict__ x, const float* __restrict__ ew,
                        const float* __restrict__ eb, float* __restrict__ xyz,
                        h16* __restrict__ feat0) {
  const int p = blockIdx.x * 256 + threadIdx.x;
  if (p >= B_ * N_) return;
  float xi[6];
#pragma unroll
  for (int i = 0; i < 6; ++i) xi[i] = x[(size_t)p * 6 + i];
  xyz[(size_t)p * 3 + 0] = xi[0];
  xyz[(size_t)p * 3 + 1] = xi[1];
  xyz[(size_t)p * 3 + 2] = xi[2];
#pragma unroll 4
  for (int o = 0; o < 64; ++o) {
    float acc = eb[o];
#pragma unroll
    for (int i = 0; i < 6; ++i) acc += xi[i] * ew[i * 64 + o];
    feat0[(size_t)p * 64 + o] = (h16)fmaxf(acc, 0.f);
  }
}

__global__ void k_gather_xyz(const float* __restrict__ src, const int* __restrict__ idx,
                             float* __restrict__ dst, int S, int NSRC) {
  int t = blockIdx.x * 256 + threadIdx.x;
  if (t >= B_ * S) return;
  int b  = t / S;
  int id = idx[t];
  const float* s = src + ((size_t)b * NSRC + id) * 3;
  dst[(size_t)t * 3 + 0] = s[0];
  dst[(size_t)t * 3 + 1] = s[1];
  dst[(size_t)t * 3 + 2] = s[2];
}

// ---------------------------------------------------------------------------
// kNN k=32: 1 thread per query; LDS point tiles; per-thread top-32 list in LDS
// (unsorted top-k is sufficient: consumers are max-pool / symmetric sums)
// ---------------------------------------------------------------------------
__global__ __launch_bounds__(128) void k_knn32(const float* __restrict__ pxyz,
                                               const float* __restrict__ qxyz,
                                               int* __restrict__ knn_out,
                                               int NQ, int NP) {
  constexpr int TP = 512;
  __shared__ float sx[TP], sy[TP], sz[TP];
  __shared__ float sD[128 * 33];
  __shared__ int   sI[128 * 33];
  const int tid = threadIdx.x;
  const int qi  = blockIdx.x * 128 + tid;
  const int b   = (blockIdx.x * 128) / NQ;   // uniform per block (NQ multiple of 128)
  const int sq  = qi - b * NQ;
  const float* q = qxyz + (size_t)(b * NQ + sq) * 3;
  const float qx = q[0], qy = q[1], qz = q[2];
  const float* P = pxyz + (size_t)b * NP * 3;
  float* D = sD + tid * 33;
  int*   I = sI + tid * 33;
  int cnt = 0, maxpos = 0;
  float dmax = -1.f;
  for (int t0 = 0; t0 < NP; t0 += TP) {
    __syncthreads();
    for (int j = tid; j < TP; j += 128) {
      sx[j] = P[(size_t)(t0 + j) * 3 + 0];
      sy[j] = P[(size_t)(t0 + j) * 3 + 1];
      sz[j] = P[(size_t)(t0 + j) * 3 + 2];
    }
    if (t0 + TP < NP) __builtin_prefetch(&P[(size_t)(t0 + TP) * 3 + tid * 12], 0, 1);
    __syncthreads();
    for (int j = 0; j < TP; ++j) {
      float dx = qx - sx[j], dy = qy - sy[j], dz = qz - sz[j];
      float d2 = dx * dx + dy * dy + dz * dz;
      if (cnt < KNN_) {
        D[cnt] = d2; I[cnt] = t0 + j;
        if (d2 > dmax) { dmax = d2; maxpos = cnt; }
        ++cnt;
      } else if (d2 < dmax) {
        D[maxpos] = d2; I[maxpos] = t0 + j;
        dmax = D[0]; maxpos = 0;
#pragma unroll
        for (int u = 1; u < KNN_; ++u)
          if (D[u] > dmax) { dmax = D[u]; maxpos = u; }
      }
    }
  }
  for (int u = 0; u < KNN_; ++u)
    knn_out[(size_t)(b * NQ + sq) * KNN_ + u] = I[u];
}

// ---------------------------------------------------------------------------
// kNN k=3 + normalized inverse-distance weights (for feature propagation)
// ---------------------------------------------------------------------------
__global__ void k_knn3(const float* __restrict__ qxyz, const float* __restrict__ pxyz,
                       int* __restrict__ idx_out, float* __restrict__ w_out,
                       int NQ, int NP) {
  const int qi = blockIdx.x * 256 + threadIdx.x;
  if (qi >= B_ * NQ) return;
  const int b = qi / NQ;
  const float qx = qxyz[(size_t)qi * 3], qy = qxyz[(size_t)qi * 3 + 1], qz = qxyz[(size_t)qi * 3 + 2];
  const float* P = pxyz + (size_t)b * NP * 3;
  float d0 = 1e30f, d1 = 1e30f, d2v = 1e30f;
  int i0 = 0, i1 = 0, i2 = 0;
  for (int j = 0; j < NP; ++j) {
    float dx = qx - P[(size_t)j * 3], dy = qy - P[(size_t)j * 3 + 1], dz = qz - P[(size_t)j * 3 + 2];
    float d = dx * dx + dy * dy + dz * dz;
    if (d < d0)       { d2v = d1; i2 = i1; d1 = d0; i1 = i0; d0 = d; i0 = j; }
    else if (d < d1)  { d2v = d1; i2 = i1; d1 = d;  i1 = j; }
    else if (d < d2v) { d2v = d;  i2 = j; }
  }
  float w0 = 1.f / fmaxf(sqrtf(fmaxf(d0, 0.f)), 1e-10f);
  float w1 = 1.f / fmaxf(sqrtf(fmaxf(d1, 0.f)), 1e-10f);
  float w2 = 1.f / fmaxf(sqrtf(fmaxf(d2v, 0.f)), 1e-10f);
  float ws = w0 + w1 + w2;
  idx_out[(size_t)qi * 3] = i0; idx_out[(size_t)qi * 3 + 1] = i1; idx_out[(size_t)qi * 3 + 2] = i2;
  w_out[(size_t)qi * 3] = w0 / ws; w_out[(size_t)qi * 3 + 1] = w1 / ws; w_out[(size_t)qi * 3 + 2] = w2 / ws;
}

// ---------------------------------------------------------------------------
// Set-abstraction MLP: rows = (b,s,kk); g=[rel(3)|feat(CF)|pad] -> CH -> CH,
// then max-pool over kk (32). 128 thr = 4 waves, 64 rows/WG (2 query groups).
// ---------------------------------------------------------------------------
template <int CF, int KP, int CH>
__global__ __launch_bounds__(128) void k_sa_mlp(
    const float* __restrict__ pxyz, const float* __restrict__ qxyz,
    const h16* __restrict__ pfeat, const int* __restrict__ knn,
    const h16* __restrict__ W1, const float* __restrict__ b1,
    const h16* __restrict__ W2, const float* __restrict__ b2,
    h16* __restrict__ out, int S, int NP) {
  extern __shared__ __attribute__((aligned(32))) h16 smem[];
  h16* sA  = smem;             // 64*KP  (swizzled, KP*16 per wave)
  h16* sH  = sA + 64 * KP;     // 64*CH  (swizzled, CH*16 per wave)
  h16* sH2 = sH + 64 * CH;     // 64*CH  (row-major)
  const int wave = threadIdx.x >> 5, lane = threadIdx.x & 31;
  const int row0 = blockIdx.x * 64;
  const int m   = lane & 15;
  const int gr  = row0 + wave * 16 + m;
  const int b   = gr / (S * KNN_);
  const int rem = gr % (S * KNN_);
  const int s = rem / KNN_, kk = rem % KNN_;
  const int pid = knn[(size_t)(b * S + s) * KNN_ + kk];
  const float* q = qxyz + (size_t)(b * S + s) * 3;
  const float* p = pxyz + ((size_t)b * NP + pid) * 3;
  const h16*   f = pfeat + ((size_t)b * NP + pid) * CF;
  h16* sAw = sA + (size_t)wave * KP * 16;
  for (int c = (lane >> 4); c < KP; c += 2) {
    h16 v;
    if (c < 3)            v = (h16)(p[c] - q[c]);
    else if (c < 3 + CF)  v = f[c - 3];
    else                  v = (h16)0.f;
    stage_swz(sAw, m, c, v);
  }
  __syncthreads();
  wave_gemm_relu<KP, CH, true>(sAw, W1, b1, sH + (size_t)wave * CH * 16);
  __syncthreads();
  wave_gemm_relu<CH, CH, false>(sH + (size_t)wave * CH * 16, W2, b2, sH2 + wave * 16 * CH);
  __syncthreads();
  for (int o = threadIdx.x; o < 2 * CH; o += 128) {
    const int g = o / CH, c = o % CH;
    float mx = 0.f; // post-ReLU values are >= 0
    for (int rr = 0; rr < KNN_; ++rr) mx = fmaxf(mx, (float)sH2[(g * KNN_ + rr) * CH + c]);
    out[(size_t)(blockIdx.x * 2 + g) * CH + c] = (h16)mx;
  }
}

// ---------------------------------------------------------------------------
// Feature propagation MLP: rows = (b,s); A=[fskip(CSKIP)|interp3(CF)] -> 128 -> 128
// ---------------------------------------------------------------------------
template <int CSKIP, int CF, int KP>
__global__ __launch_bounds__(128) void k_fp_mlp(
    const h16* __restrict__ fskip, const h16* __restrict__ fsrc,
    const int* __restrict__ idx3, const float* __restrict__ w3,
    const h16* __restrict__ W1, const float* __restrict__ b1,
    const h16* __restrict__ W2, const float* __restrict__ b2,
    h16* __restrict__ out, int S, int SRCS) {
  constexpr int CH = 128;
  extern __shared__ __attribute__((aligned(32))) h16 smem[];
  h16* sA  = smem;             // 64*KP (swizzled)
  h16* sH  = sA + 64 * KP;     // 64*CH (swizzled)
  h16* sH2 = sH + 64 * CH;     // 64*CH (row-major)
  const int wave = threadIdx.x >> 5, lane = threadIdx.x & 31;
  const int row0 = blockIdx.x * 64;
  const int m  = lane & 15;
  const int gr = row0 + wave * 16 + m;
  const int b  = gr / S;
  const size_t base = (size_t)gr * 3;
  const int   i0 = idx3[base], i1 = idx3[base + 1], i2 = idx3[base + 2];
  const float w0 = w3[base], w1 = w3[base + 1], w2 = w3[base + 2];
  const h16* fs = fskip + (size_t)gr * CSKIP;
  const h16* f0 = fsrc + ((size_t)b * SRCS + i0) * CF;
  const h16* f1 = fsrc + ((size_t)b * SRCS + i1) * CF;
  const h16* f2 = fsrc + ((size_t)b * SRCS + i2) * CF;
  h16* sAw = sA + (size_t)wave * KP * 16;
  for (int c = (lane >> 4); c < KP; c += 2) {   // KP == CSKIP + CF
    h16 v;
    if (c < CSKIP) v = fs[c];
    else {
      int cc = c - CSKIP;
      v = (h16)(w0 * (float)f0[cc] + w1 * (float)f1[cc] + w2 * (float)f2[cc]);
    }
    stage_swz(sAw, m, c, v);
  }
  __syncthreads();
  wave_gemm_relu<KP, CH, true>(sAw, W1, b1, sH + (size_t)wave * CH * 16);
  __syncthreads();
  wave_gemm_relu<CH, CH, false>(sH + (size_t)wave * CH * 16, W2, b2, sH2 + wave * 16 * CH);
  __syncthreads();
  for (int o = threadIdx.x; o < 64 * CH; o += 128) {
    int rr = o / CH, c = o % CH;
    out[(size_t)(row0 + rr) * CH + c] = sH2[rr * CH + c];
  }
}

// ---------------------------------------------------------------------------
// Global max over N of feat0_up (post-ReLU, >=0) -> gf (B,128) f32
// ---------------------------------------------------------------------------
__global__ __launch_bounds__(128) void k_gmax(const h16* __restrict__ feat, float* __restrict__ gf) {
  const int b = blockIdx.x, c = threadIdx.x;
  float m = 0.f;
  for (int r = 0; r < N_; ++r) m = fmaxf(m, (float)feat[((size_t)b * N_ + r) * 128 + c]);
  gf[b * 128 + c] = m;
}

// ---------------------------------------------------------------------------
// Head: A=[feat0_up(128)|gf(128)] -> 128 (WMMA) -> 64 (WMMA) -> 13 (VALU)
// ---------------------------------------------------------------------------
__global__ __launch_bounds__(128) void k_head(
    const h16* __restrict__ feat, const float* __restrict__ gf,
    const h16* __restrict__ W1, const float* __restrict__ b1,
    const h16* __restrict__ W2, const float* __restrict__ b2,
    const float* __restrict__ W3, const float* __restrict__ b3,
    float* __restrict__ out) {
  extern __shared__ __attribute__((aligned(32))) h16 smem[];
  h16* sA  = smem;             // 64*256 (swizzled)
  h16* sH  = sA + 64 * 256;    // 64*128 (swizzled)
  h16* sH2 = sH + 64 * 128;    // 64*64  (row-major)
  const int wave = threadIdx.x >> 5, lane = threadIdx.x & 31;
  const int row0 = blockIdx.x * 64;
  const int m  = lane & 15;
  const int gr = row0 + wave * 16 + m;
  const int b  = gr / N_;
  const h16* fr = feat + (size_t)gr * 128;
  const float* g = gf + b * 128;
  h16* sAw = sA + (size_t)wave * 256 * 16;
  for (int c = (lane >> 4); c < 256; c += 2)
    stage_swz(sAw, m, c, (c < 128) ? fr[c] : (h16)g[c - 128]);
  __syncthreads();
  wave_gemm_relu<256, 128, true>(sAw, W1, b1, sH + (size_t)wave * 128 * 16);
  __syncthreads();
  wave_gemm_relu<128, 64, false>(sH + (size_t)wave * 128 * 16, W2, b2, sH2 + wave * 16 * 64);
  __syncthreads();
  for (int o = threadIdx.x; o < 64 * NCLS_; o += 128) {
    int rr = o / NCLS_, cls = o % NCLS_;
    float acc = b3[cls];
#pragma unroll 8
    for (int k = 0; k < 64; ++k) acc += (float)sH2[rr * 64 + k] * W3[k * NCLS_ + cls];
    out[(size_t)(row0 + rr) * NCLS_ + cls] = acc;
  }
}

// ---------------------------------------------------------------------------
extern "C" void kernel_launch(void* const* d_in, const int* in_sizes, int n_in,
                              void* d_out, int out_size, void* d_ws, size_t ws_size,
                              hipStream_t stream) {
  (void)in_sizes; (void)n_in; (void)out_size; (void)ws_size;
  const float* x       = (const float*)d_in[0];
  const int*   idx_s1  = (const int*)d_in[1];
  const int*   idx_s2  = (const int*)d_in[2];
  const float* ew      = (const float*)d_in[3];
  const float* ebias   = (const float*)d_in[4];
  const float* sa1w1   = (const float*)d_in[5];
  const float* sa1b1   = (const float*)d_in[6];
  const float* sa1w2   = (const float*)d_in[7];
  const float* sa1b2   = (const float*)d_in[8];
  const float* sa2w1   = (const float*)d_in[9];
  const float* sa2b1   = (const float*)d_in[10];
  const float* sa2w2   = (const float*)d_in[11];
  const float* sa2b2   = (const float*)d_in[12];
  const float* fp2w1   = (const float*)d_in[13];
  const float* fp2b1   = (const float*)d_in[14];
  const float* fp2w2   = (const float*)d_in[15];
  const float* fp2b2   = (const float*)d_in[16];
  const float* fp1w1   = (const float*)d_in[17];
  const float* fp1b1   = (const float*)d_in[18];
  const float* fp1w2   = (const float*)d_in[19];
  const float* fp1b2   = (const float*)d_in[20];
  const float* hw1     = (const float*)d_in[21];
  const float* hb1     = (const float*)d_in[22];
  const float* hw2     = (const float*)d_in[23];
  const float* hb2     = (const float*)d_in[24];
  const float* hw3     = (const float*)d_in[25];
  const float* hb3     = (const float*)d_in[26];
  float* outp = (float*)d_out;

  // ---- workspace bump allocator (256B aligned regions) ----
  size_t off = 0;
  auto alloc = [&](size_t bytes) -> char* {
    char* p = (char*)d_ws + off;
    off += (bytes + 255) & ~(size_t)255;
    return p;
  };
  float* xyz      = (float*)alloc((size_t)B_ * N_ * 3 * 4);
  float* xyz1     = (float*)alloc((size_t)B_ * S1_ * 3 * 4);
  float* xyz2     = (float*)alloc((size_t)B_ * S2_ * 3 * 4);
  h16*   feat0    = (h16*)alloc((size_t)B_ * N_ * 64 * 2);
  h16*   feat1    = (h16*)alloc((size_t)B_ * S1_ * 128 * 2);
  h16*   feat2    = (h16*)alloc((size_t)B_ * S2_ * 256 * 2);
  h16*   feat1u   = (h16*)alloc((size_t)B_ * S1_ * 128 * 2);
  h16*   feat0u   = (h16*)alloc((size_t)B_ * N_ * 128 * 2);
  float* gfbuf    = (float*)alloc((size_t)B_ * 128 * 4);
  int*   knn1     = (int*)alloc((size_t)B_ * S1_ * KNN_ * 4);
  int*   knn2     = (int*)alloc((size_t)B_ * S2_ * KNN_ * 4);
  int*   fp2idx   = (int*)alloc((size_t)B_ * S1_ * 3 * 4);
  float* fp2w     = (float*)alloc((size_t)B_ * S1_ * 3 * 4);
  int*   fp1idx   = (int*)alloc((size_t)B_ * N_ * 3 * 4);
  float* fp1w     = (float*)alloc((size_t)B_ * N_ * 3 * 4);
  h16* sa1w1p = (h16*)alloc((size_t)96  * 128 * 2);
  h16* sa1w2p = (h16*)alloc((size_t)128 * 128 * 2);
  h16* sa2w1p = (h16*)alloc((size_t)160 * 256 * 2);
  h16* sa2w2p = (h16*)alloc((size_t)256 * 256 * 2);
  h16* fp2w1p = (h16*)alloc((size_t)384 * 128 * 2);
  h16* fp2w2p = (h16*)alloc((size_t)128 * 128 * 2);
  h16* fp1w1p = (h16*)alloc((size_t)192 * 128 * 2);
  h16* fp1w2p = (h16*)alloc((size_t)128 * 128 * 2);
  h16* hw1p   = (h16*)alloc((size_t)256 * 128 * 2);
  h16* hw2p   = (h16*)alloc((size_t)128 * 64 * 2);

  auto cvt = [&](const float* w, h16* o, int K, int N, int Kp) {
    k_cvt_pack<<<(Kp * N + 255) / 256, 256, 0, stream>>>(w, o, K, N, Kp);
  };
  cvt(sa1w1, sa1w1p, 67, 128, 96);
  cvt(sa1w2, sa1w2p, 128, 128, 128);
  cvt(sa2w1, sa2w1p, 131, 256, 160);
  cvt(sa2w2, sa2w2p, 256, 256, 256);
  cvt(fp2w1, fp2w1p, 384, 128, 384);
  cvt(fp2w2, fp2w2p, 128, 128, 128);
  cvt(fp1w1, fp1w1p, 192, 128, 192);
  cvt(fp1w2, fp1w2p, 128, 128, 128);
  cvt(hw1, hw1p, 256, 128, 256);
  cvt(hw2, hw2p, 128, 64, 128);

  // embed + xyz extract
  k_embed<<<(B_ * N_ + 255) / 256, 256, 0, stream>>>(x, ew, ebias, xyz, feat0);

  // ---- SA1 ----
  k_gather_xyz<<<(B_ * S1_ + 255) / 256, 256, 0, stream>>>(xyz, idx_s1, xyz1, S1_, N_);
  k_knn32<<<B_ * S1_ / 128, 128, 0, stream>>>(xyz, xyz1, knn1, S1_, N_);
  {
    constexpr size_t sh = (size_t)(64 * 96 + 2 * 64 * 128) * 2;
    k_sa_mlp<64, 96, 128><<<B_ * S1_ * KNN_ / 64, 128, sh, stream>>>(
        xyz, xyz1, feat0, knn1, sa1w1p, sa1b1, sa1w2p, sa1b2, feat1, S1_, N_);
  }

  // ---- SA2 ----
  k_gather_xyz<<<(B_ * S2_ + 255) / 256, 256, 0, stream>>>(xyz1, idx_s2, xyz2, S2_, S1_);
  k_knn32<<<B_ * S2_ / 128, 128, 0, stream>>>(xyz1, xyz2, knn2, S2_, S1_);
  {
    constexpr size_t sh = (size_t)(64 * 160 + 2 * 64 * 256) * 2;
    k_sa_mlp<128, 160, 256><<<B_ * S2_ * KNN_ / 64, 128, sh, stream>>>(
        xyz1, xyz2, feat1, knn2, sa2w1p, sa2b1, sa2w2p, sa2b2, feat2, S2_, S1_);
  }

  // ---- FP2: (xyz1 <- xyz2) ----
  k_knn3<<<(B_ * S1_ + 255) / 256, 256, 0, stream>>>(xyz1, xyz2, fp2idx, fp2w, S1_, S2_);
  {
    constexpr size_t sh = (size_t)(64 * 384 + 2 * 64 * 128) * 2;
    k_fp_mlp<128, 256, 384><<<B_ * S1_ / 64, 128, sh, stream>>>(
        feat1, feat2, fp2idx, fp2w, fp2w1p, fp2b1, fp2w2p, fp2b2, feat1u, S1_, S2_);
  }

  // ---- FP1: (xyz <- xyz1) ----
  k_knn3<<<(B_ * N_ + 255) / 256, 256, 0, stream>>>(xyz, xyz1, fp1idx, fp1w, N_, S1_);
  {
    constexpr size_t sh = (size_t)(64 * 192 + 2 * 64 * 128) * 2;
    k_fp_mlp<64, 128, 192><<<B_ * N_ / 64, 128, sh, stream>>>(
        feat0, feat1u, fp1idx, fp1w, fp1w1p, fp1b1, fp1w2p, fp1b2, feat0u, N_, S1_);
  }

  // ---- global max + head ----
  k_gmax<<<B_, 128, 0, stream>>>(feat0u, gfbuf);
  {
    constexpr size_t sh = (size_t)(64 * 256 + 64 * 128 + 64 * 64) * 2;
    k_head<<<B_ * N_ / 64, 128, sh, stream>>>(
        feat0u, gfbuf, hw1p, hb1, hw2p, hb2, hw3, hb3, outp);
  }
}
